// MultiHeadAttention_45337674776859
// MI455X (gfx1250) — matrix operand order repaired and checked by reading, not compiled, not run on hardware
//
#include <hip/hip_runtime.h>
#include <hip/hip_bf16.h>

#define EMBED 1024
#define HEADS 16
#define HDIM  64
#define BATCH 2
#define SEQ   2048
#define ROWS  (BATCH * SEQ)   // 4096
#define KBLK  64
#define LDP   72              // padded LDS row pitch (halfs) from TDM pad feature

typedef _Float16 h16v __attribute__((ext_vector_type(16)));
typedef _Float16 h8v  __attribute__((ext_vector_type(8)));
typedef float    f8v  __attribute__((ext_vector_type(8)));
typedef unsigned int u32x4 __attribute__((ext_vector_type(4)));
typedef unsigned int u32x8 __attribute__((ext_vector_type(8)));

union FragU { h16v v; h8v h[2]; };

// ---- WMMA fragment loaders (layouts per CDNA5 ISA 7.12.2) -----------------

// A (16x32 f16, MxK), row-major source with leading dim ld.
__device__ __forceinline__ h16v load_frag_a(const _Float16* __restrict__ base,
                                            int ld, int lane) {
  const int row  = lane & 15;
  const int koff = (lane >> 4) << 3;
  const _Float16* p = base + row * ld + koff;
  FragU f;
  f.h[0] = *(const h8v*)(p);
  f.h[1] = *(const h8v*)(p + 16);
  return f.v;
}

// B (32x16 f16, KxN) supplied as B^T row-major [N][K] with leading dim ld.
__device__ __forceinline__ h16v load_frag_b(const _Float16* __restrict__ bt,
                                            int ld, int lane) {
  const int n    = lane & 15;
  const int koff = (lane >> 4) << 3;
  const _Float16* p = bt + n * ld + koff;
  FragU f;
  f.h[0] = *(const h8v*)(p);
  f.h[1] = *(const h8v*)(p + 16);
  return f.v;
}

#define WMMA_F16(A, B, C) \
  __builtin_amdgcn_wmma_f32_16x16x32_f16(false, (A), false, (B), (short)0, (C), false, false)

// ---- Tensor Data Mover: DMA a 64x64 f16 tile (row stride `stride_elems`)
// from global into LDS at byte offset lds_off, padding each 64-half row to
// 72 halfs in LDS (pad_interval=32 DWORDs -> code 4, pad_amount=4 DWORDs ->
// code 3). Descriptor layout per CDNA5 ISA 8.3/8.4. Issued wave-uniform.
__device__ __forceinline__ void tdm_load_tile64(const _Float16* gptr,
                                                unsigned int lds_off,
                                                unsigned int stride_elems) {
  const unsigned long long ga = (unsigned long long)(size_t)gptr;
  u32x4 g0;
  g0[0] = 1u;                                              // count=1, user mode
  g0[1] = lds_off;                                         // lds_addr (bytes)
  g0[2] = (unsigned int)ga;                                // global_addr[31:0]
  g0[3] = ((unsigned int)(ga >> 32) & 0x01FFFFFFu)         // global_addr[56:32]
          | 0x80000000u;                                   // type=2 ("image")
  u32x8 g1;
  g1[0] = (1u << 16)        // data_size = 2 bytes
        | (1u << 20)        // pad_enable
        | (4u << 22)        // pad_interval: 32 DWORDs (one 64-half row)
        | (3u << 25);       // pad_amount:   4 DWORDs (8 halfs) -> pitch 72
  g1[1] = 64u << 16;        // tensor_dim0 = 64
  g1[2] = 64u << 16;        // tensor_dim1 = 64
  g1[3] = 64u << 16;        // tile_dim0   = 64
  g1[4] = 64u;              // tile_dim1   = 64, tile_dim2 = 0
  g1[5] = stride_elems;     // tensor_dim0_stride[31:0]
  g1[6] = 0u;
  g1[7] = 0u;
  asm volatile("tensor_load_to_lds %0, %1" :: "s"(g0), "s"(g1) : "memory");
}

// ---- elementwise converters ------------------------------------------------

__global__ void mha_cvt_f16(const float* __restrict__ in,
                            _Float16* __restrict__ out, int n) {
  int i = blockIdx.x * blockDim.x + threadIdx.x;
  if (i < n) out[i] = (_Float16)in[i];
}

// Wt[n][k] = (f16) W[k][n]   (EMBED x EMBED)
__global__ void mha_trans_cvt(const float* __restrict__ in,
                              _Float16* __restrict__ out) {
  int i = blockIdx.x * blockDim.x + threadIdx.x;
  int k = i >> 10;
  int n = i & (EMBED - 1);
  out[(size_t)n * EMBED + k] = (_Float16)in[(size_t)k * EMBED + n];
}

// ---- generic f16 GEMM: C = A @ Bt^T + bias ---------------------------------
// Wave tile 32x64 (2x4 WMMA tiles); workgroup tile 64x256 (8 waves, 2x4).
// mode 0/1: store f16 into head-split [B,H,S,64] (mode 0 = Q, scaled)
// mode 2:   store f16 transposed   [B,H,64,S]   (V)
// mode 3:   store f32 row-major    [M,N]        (output projection)
__global__ __launch_bounds__(256)
void mha_gemm_f16(const _Float16* __restrict__ A, const _Float16* __restrict__ Bt,
                  const float* __restrict__ bias, void* __restrict__ outp,
                  int M, int N, int K, int mode, float scale) {
  const int lane = threadIdx.x & 31;
  const int wave = threadIdx.x >> 5;
  const int wm = wave >> 2, wn = wave & 3;
  const int m0 = blockIdx.y * 64 + wm * 32;
  const int n0 = blockIdx.x * 256 + wn * 64;

  f8v acc[2][4] = {};
  for (int k0 = 0; k0 < K; k0 += 32) {
    h16v a0 = load_frag_a(A + (size_t)m0 * K + k0, K, lane);
    h16v a1 = load_frag_a(A + (size_t)(m0 + 16) * K + k0, K, lane);
    h16v b0 = load_frag_b(Bt + (size_t)(n0 +  0) * K + k0, K, lane);
    h16v b1 = load_frag_b(Bt + (size_t)(n0 + 16) * K + k0, K, lane);
    h16v b2 = load_frag_b(Bt + (size_t)(n0 + 32) * K + k0, K, lane);
    h16v b3 = load_frag_b(Bt + (size_t)(n0 + 48) * K + k0, K, lane);
    acc[0][0] = WMMA_F16(a0, b0, acc[0][0]);
    acc[0][1] = WMMA_F16(a0, b1, acc[0][1]);
    acc[0][2] = WMMA_F16(a0, b2, acc[0][2]);
    acc[0][3] = WMMA_F16(a0, b3, acc[0][3]);
    acc[1][0] = WMMA_F16(a1, b0, acc[1][0]);
    acc[1][1] = WMMA_F16(a1, b1, acc[1][1]);
    acc[1][2] = WMMA_F16(a1, b2, acc[1][2]);
    acc[1][3] = WMMA_F16(a1, b3, acc[1][3]);
  }

  const int rsel = (lane >> 4) << 3;
  const int csel = lane & 15;
#pragma unroll
  for (int tm = 0; tm < 2; ++tm)
#pragma unroll
    for (int tn = 0; tn < 4; ++tn)
#pragma unroll
      for (int r = 0; r < 8; ++r) {
        const int row = m0 + tm * 16 + r + rsel;
        const int col = n0 + tn * 16 + csel;
        const float v = (acc[tm][tn][r] + bias[col]) * scale;
        if (mode == 3) {
          ((float*)outp)[(size_t)row * N + col] = v;
        } else {
          const int bb = row >> 11, ss = row & (SEQ - 1);
          const int hh = col >> 6,  dd = col & (HDIM - 1);
          _Float16* o16 = (_Float16*)outp;
          if (mode == 2)
            o16[((size_t)(bb * HEADS + hh) * HDIM + dd) * SEQ + ss] = (_Float16)v;
          else
            o16[((size_t)(bb * HEADS + hh) * SEQ + ss) * HDIM + dd] = (_Float16)v;
        }
      }
}

// ---- fused flash attention -------------------------------------------------
// grid.x = B*H (32), grid.y = S/128 (16); 256 threads = 8 waves, each wave
// owns 16 query rows. K/V 64x64 tiles are DMA'd into LDS by the TDM
// (double-buffered, overlapped with compute); fragments read via ds_load.
__global__ __launch_bounds__(256)
void mha_flash(const _Float16* __restrict__ Qp, const _Float16* __restrict__ Kp,
               const _Float16* __restrict__ Vt, _Float16* __restrict__ ctx) {
  __shared__ _Float16 smK[2][KBLK * LDP];   // padded 72-half pitch, 2x9 KB
  __shared__ _Float16 smV[2][KBLK * LDP];   // padded 72-half pitch, 2x9 KB
  __shared__ _Float16 Plds[8][16 * LDP];    // per-wave P staging, 18 KB

  const int lane = threadIdx.x & 31;
  const int wave = threadIdx.x >> 5;
  const int bh = blockIdx.x;
  const int b = bh >> 4, h = bh & 15;
  const _Float16* Qb = Qp + (size_t)bh * SEQ * HDIM;
  const _Float16* Kb = Kp + (size_t)bh * SEQ * HDIM;
  const _Float16* Vb = Vt + (size_t)bh * HDIM * SEQ;

  // LDS byte offsets (flat LDS-aperture address low 32 bits == LDS offset)
  const unsigned int kOff[2] = {(unsigned int)(size_t)&smK[0][0],
                                (unsigned int)(size_t)&smK[1][0]};
  const unsigned int vOff[2] = {(unsigned int)(size_t)&smV[0][0],
                                (unsigned int)(size_t)&smV[1][0]};

  const int q0 = blockIdx.y * 128 + wave * 16;

  // Q fragments (Dh=64 -> two K=32 steps), kept in registers for all blocks
  h16v qa[2];
  qa[0] = load_frag_a(Qb + (size_t)q0 * HDIM + 0, HDIM, lane);
  qa[1] = load_frag_a(Qb + (size_t)q0 * HDIM + 32, HDIM, lane);

  f8v o[4] = {};
  float m[8], l[8];
#pragma unroll
  for (int r = 0; r < 8; ++r) { m[r] = -__builtin_inff(); l[r] = 0.0f; }

  const int prow = (lane >> 4) << 3;
  const int pcol = lane & 15;

  // Prefetch first K/V tiles via TDM (wave 0 issues; instruction is wave-level)
  if (wave == 0) {
    tdm_load_tile64(Kb, kOff[0], HDIM);   // K tile: rows=keys, stride 64
    tdm_load_tile64(Vb, vOff[0], SEQ);    // V tile: rows=dh,   stride 2048
  }

  for (int i = 0; i < SEQ / KBLK; ++i) {
    const int cur = i & 1;
    if (wave == 0) __builtin_amdgcn_s_wait_tensorcnt(0);
    __syncthreads();   // publish TDM-written tiles to all waves

    // kick off DMA of the next block into the other buffer (overlaps compute)
    if (wave == 0 && (i + 1) < SEQ / KBLK) {
      const int nb = (i + 1) & 1;
      const int kb2 = (i + 1) * KBLK;
      tdm_load_tile64(Kb + (size_t)kb2 * HDIM, kOff[nb], HDIM);
      tdm_load_tile64(Vb + kb2, vOff[nb], SEQ);
    }

    const _Float16* Kl = &smK[cur][0];
    const _Float16* Vl = &smV[cur][0];

    // scores tile 16x64 (Q pre-scaled by 1/sqrt(Dh) at projection)
    f8v sc[4] = {};
#pragma unroll
    for (int nt = 0; nt < 4; ++nt) {
      h16v kf0 = load_frag_b(Kl + (nt * 16) * LDP + 0, LDP, lane);
      h16v kf1 = load_frag_b(Kl + (nt * 16) * LDP + 32, LDP, lane);
      sc[nt] = WMMA_F16(qa[0], kf0, sc[nt]);
      sc[nt] = WMMA_F16(qa[1], kf1, sc[nt]);
    }

    // online softmax: row r lives in VGPR r of one 16-lane half-wave
#pragma unroll
    for (int r = 0; r < 8; ++r) {
      float mx = fmaxf(fmaxf(sc[0][r], sc[1][r]), fmaxf(sc[2][r], sc[3][r]));
#pragma unroll
      for (int off = 1; off < 16; off <<= 1)
        mx = fmaxf(mx, __shfl_xor(mx, off, 32));
      const float mn = fmaxf(m[r], mx);
      const float alpha = __expf(m[r] - mn);
      m[r] = mn;
      const float p0 = __expf(sc[0][r] - mn);
      const float p1 = __expf(sc[1][r] - mn);
      const float p2 = __expf(sc[2][r] - mn);
      const float p3 = __expf(sc[3][r] - mn);
      sc[0][r] = p0; sc[1][r] = p1; sc[2][r] = p2; sc[3][r] = p3;
      float s = p0 + p1 + p2 + p3;
#pragma unroll
      for (int off = 1; off < 16; off <<= 1)
        s += __shfl_xor(s, off, 32);
      l[r] = l[r] * alpha + s;
      o[0][r] *= alpha; o[1][r] *= alpha; o[2][r] *= alpha; o[3][r] *= alpha;
    }

    // re-layout P: D-fragment (col-per-lane) -> row-major LDS tile (per wave)
    __syncthreads();
#pragma unroll
    for (int nt = 0; nt < 4; ++nt)
#pragma unroll
      for (int r = 0; r < 8; ++r)
        Plds[wave][(r + prow) * LDP + nt * 16 + pcol] = (_Float16)sc[nt][r];
    __syncthreads();

    // P A-fragments from LDS (two K=32 steps over the 64 keys)
    h16v pa[2];
    {
      const int row = lane & 15;
      const int koff = (lane >> 4) << 3;
#pragma unroll
      for (int ks = 0; ks < 2; ++ks) {
        const _Float16* pp = &Plds[wave][row * LDP + ks * 32 + koff];
        FragU f;
        f.h[0] = *(const h8v*)(pp);
        f.h[1] = *(const h8v*)(pp + 16);
        pa[ks] = f.v;
      }
    }

    // ctx += P @ V  (V tile in LDS is [dh][key], pitch 72)
#pragma unroll
    for (int nt = 0; nt < 4; ++nt) {
      h16v vf0 = load_frag_b(Vl + (nt * 16) * LDP + 0, LDP, lane);
      h16v vf1 = load_frag_b(Vl + (nt * 16) * LDP + 32, LDP, lane);
      o[nt] = WMMA_F16(pa[0], vf0, o[nt]);
      o[nt] = WMMA_F16(pa[1], vf1, o[nt]);
    }
    __syncthreads();   // all waves done with `cur` before TDM reuses it
  }

  // normalize and store ctx as f16 in [B,S,E] layout
#pragma unroll
  for (int r = 0; r < 8; ++r) {
    const float inv = 1.0f / l[r];
    const int q = q0 + r + prow;
    const size_t base = ((size_t)(b * SEQ + q)) * EMBED + h * HDIM;
#pragma unroll
    for (int nt = 0; nt < 4; ++nt)
      ctx[base + nt * 16 + pcol] = (_Float16)(o[nt][r] * inv);
  }
}

// ---- host side -------------------------------------------------------------

extern "C" void kernel_launch(void* const* d_in, const int* in_sizes, int n_in,
                              void* d_out, int out_size, void* d_ws, size_t ws_size,
                              hipStream_t stream) {
  const float* q  = (const float*)d_in[0];
  const float* k  = (const float*)d_in[1];
  const float* v  = (const float*)d_in[2];
  const float* Wq = (const float*)d_in[3];
  const float* bq = (const float*)d_in[4];
  const float* Wk = (const float*)d_in[5];
  const float* bk = (const float*)d_in[6];
  const float* Wv = (const float*)d_in[7];
  const float* bv = (const float*)d_in[8];
  const float* Wo = (const float*)d_in[9];
  const float* bo = (const float*)d_in[10];
  float* out = (float*)d_out;

  const size_t NX = (size_t)ROWS * EMBED;   // 4M elements
  const size_t NW = (size_t)EMBED * EMBED;  // 1M elements
  _Float16* w = (_Float16*)d_ws;
  _Float16* Xq  = w; w += NX;
  _Float16* Xk  = w; w += NX;
  _Float16* Xv  = w; w += NX;
  _Float16* Wqt = w; w += NW;
  _Float16* Wkt = w; w += NW;
  _Float16* Wvt = w; w += NW;
  _Float16* Wot = w; w += NW;
  _Float16* Qp  = w; w += NX;   // [B,H,S,64], pre-scaled by 1/8
  _Float16* Kp  = w; w += NX;   // [B,H,S,64]
  _Float16* Vtp = w; w += NX;   // [B,H,64,S]
  _Float16* Cx  = w; w += NX;   // [B,S,E]

  mha_cvt_f16<<<NX / 256, 256, 0, stream>>>(q, Xq, (int)NX);
  mha_cvt_f16<<<NX / 256, 256, 0, stream>>>(k, Xk, (int)NX);
  mha_cvt_f16<<<NX / 256, 256, 0, stream>>>(v, Xv, (int)NX);
  mha_trans_cvt<<<NW / 256, 256, 0, stream>>>(Wq, Wqt);
  mha_trans_cvt<<<NW / 256, 256, 0, stream>>>(Wk, Wkt);
  mha_trans_cvt<<<NW / 256, 256, 0, stream>>>(Wv, Wvt);
  mha_trans_cvt<<<NW / 256, 256, 0, stream>>>(Wo, Wot);

  dim3 gg(EMBED / 256, ROWS / 64);   // (4, 64)
  mha_gemm_f16<<<gg, 256, 0, stream>>>(Xq, Wqt, bq, Qp,  ROWS, EMBED, EMBED, 0, 0.125f);
  mha_gemm_f16<<<gg, 256, 0, stream>>>(Xk, Wkt, bk, Kp,  ROWS, EMBED, EMBED, 1, 1.0f);
  mha_gemm_f16<<<gg, 256, 0, stream>>>(Xv, Wvt, bv, Vtp, ROWS, EMBED, EMBED, 2, 1.0f);

  mha_flash<<<dim3(BATCH * HEADS, SEQ / 128), 256, 0, stream>>>(Qp, Kp, Vtp, Cx);

  mha_gemm_f16<<<gg, 256, 0, stream>>>(Cx, Wot, bo, out, ROWS, EMBED, EMBED, 3, 1.0f);
}
